// TernaryLinear_77824807404253
// MI455X (gfx1250) — matrix operand order repaired and checked by reading, not compiled, not run on hardware
//
#include <hip/hip_runtime.h>
#include <hip/hip_bf16.h>
#include <stdint.h>

// ---------------------------------------------------------------------------
// TernaryLinear for MI455X (gfx1250):
//   y[b,s,o] = sum_i x[b,s,i] * Q(W)[o,i] + bias[o]
// Ternary weights are exact in bf16; x converted to bf16; GEMM runs on
// v_wmma_f32_16x16x32_bf16 with f32 accumulation (bias folded into acc init).
// M = 8192, K = 2048, N = 2048.
// 64x64 wave tiles (16 wmma / K-step / wave, ds:wmma = 1.0), double-buffered
// LDS with async global->LDS copies (ASYNCcnt path), one barrier per K-step.
// ---------------------------------------------------------------------------

typedef __attribute__((ext_vector_type(16))) __bf16 v16bf;
typedef __attribute__((ext_vector_type(8)))  float  v8f;
typedef __attribute__((ext_vector_type(4)))  unsigned int v4u;
typedef __attribute__((ext_vector_type(4)))  int    v4i;
typedef __attribute__((ext_vector_type(4)))  float  v4f;
typedef __attribute__((ext_vector_type(4)))  unsigned short v4s;

#define AS1 __attribute__((address_space(1)))
#define AS3 __attribute__((address_space(3)))

#if __has_builtin(__builtin_amdgcn_global_load_async_to_lds_b128)
#define HAVE_ASYNC_LDS 1
#else
#define HAVE_ASYNC_LDS 0
#endif

union Frag {
    v4u   u[2];
    v16bf v;
};

__device__ __forceinline__ uint16_t f32_to_bf16_rne(float f) {
    uint32_t u = __float_as_uint(f);
    uint32_t r = u + 0x7FFFu + ((u >> 16) & 1u);
    return (uint16_t)(r >> 16);
}

// ---------------------------------------------------------------------------
// Stage 1: per-output-row ternary quantization of W -> bf16 {-1,0,+1}
// ---------------------------------------------------------------------------
__global__ __launch_bounds__(256) void quantize_w(const float* __restrict__ W,
                                                  uint16_t* __restrict__ Wq,
                                                  int K) {
    const int row = blockIdx.x;
    const float* w = W + (size_t)row * K;

    float s = 0.0f;
    for (int i = threadIdx.x; i < K; i += 256) s += fabsf(w[i]);

    __shared__ float red[256];
    red[threadIdx.x] = s;
    __syncthreads();
    for (int off = 128; off > 0; off >>= 1) {
        if (threadIdx.x < off) red[threadIdx.x] += red[threadIdx.x + off];
        __syncthreads();
    }
    const float thr = 0.7f * red[0] / (float)K;

    uint16_t* o = Wq + (size_t)row * K;
    for (int i = threadIdx.x; i < K; i += 256) {
        float v = w[i];
        float q = (v > thr) ? 1.0f : ((v < -thr) ? -1.0f : 0.0f);
        o[i] = f32_to_bf16_rne(q);   // exact for {-1,0,1}
    }
}

// ---------------------------------------------------------------------------
// Stage 2: x f32 -> bf16 (RNE), 4 elements / thread
// ---------------------------------------------------------------------------
__global__ __launch_bounds__(256) void convert_x(const float* __restrict__ X,
                                                 uint16_t* __restrict__ Xh,
                                                 long long n4) {
    long long i = (long long)blockIdx.x * blockDim.x + threadIdx.x;
    if (i < n4) {
        v4f v = ((const v4f*)X)[i];
        v4s o;
        o.x = f32_to_bf16_rne(v.x);
        o.y = f32_to_bf16_rne(v.y);
        o.z = f32_to_bf16_rne(v.z);
        o.w = f32_to_bf16_rne(v.w);
        ((v4s*)Xh)[i] = o;
    }
}

// ---------------------------------------------------------------------------
// Stage 3: WMMA GEMM.  C[M,N] = Xh[M,K] * Wq[N,K]^T + bias[N]
// WG tile 256x128, 8 wave32s arranged 4(M) x 2(N); each wave owns 64x64 =
// 4x4 WMMA tiles.  K-step 32, double-buffered LDS, async global->LDS copies.
// ---------------------------------------------------------------------------
#define BM 256
#define BN 128
#define BK 32

// Stage one K-slice of A (256x32) and B (128x32) bf16 tiles into LDS.
// A: 1024 16-byte chunks -> 4/thread.  B: 512 chunks -> 2/thread.
__device__ __forceinline__ void stage_tiles(const uint16_t* __restrict__ Xh,
                                            const uint16_t* __restrict__ Wq,
                                            uint16_t* As, uint16_t* Bs,
                                            int m0, int n0, int k0, int Kd,
                                            int tid) {
#if HAVE_ASYNC_LDS
#pragma unroll
    for (int j = 0; j < 4; ++j) {
        const int c   = tid + 256 * j;
        const int row = c >> 2;
        const int col = (c & 3) * 8;
        __builtin_amdgcn_global_load_async_to_lds_b128(
            (AS1 v4i*)(Xh + (size_t)(m0 + row) * Kd + k0 + col),
            (AS3 v4i*)(As + row * BK + col), 0, 0);
    }
#pragma unroll
    for (int j = 0; j < 2; ++j) {
        const int c   = tid + 256 * j;
        const int row = c >> 2;
        const int col = (c & 3) * 8;
        __builtin_amdgcn_global_load_async_to_lds_b128(
            (AS1 v4i*)(Wq + (size_t)(n0 + row) * Kd + k0 + col),
            (AS3 v4i*)(Bs + row * BK + col), 0, 0);
    }
#else
#pragma unroll
    for (int j = 0; j < 4; ++j) {
        const int c   = tid + 256 * j;
        const int row = c >> 2;
        const int col = (c & 3) * 8;
        *(v4u*)(As + row * BK + col) =
            *(const v4u*)(Xh + (size_t)(m0 + row) * Kd + k0 + col);
    }
#pragma unroll
    for (int j = 0; j < 2; ++j) {
        const int c   = tid + 256 * j;
        const int row = c >> 2;
        const int col = (c & 3) * 8;
        *(v4u*)(Bs + row * BK + col) =
            *(const v4u*)(Wq + (size_t)(n0 + row) * Kd + k0 + col);
    }
#endif
}

__device__ __forceinline__ void wait_async_copies() {
#if HAVE_ASYNC_LDS
#if __has_builtin(__builtin_amdgcn_s_wait_asynccnt)
    __builtin_amdgcn_s_wait_asynccnt(0);
#else
    asm volatile("s_wait_asynccnt 0x0" ::: "memory");
#endif
#endif
}

__global__ __launch_bounds__(256) void gemm_wmma(const uint16_t* __restrict__ Xh,
                                                 const uint16_t* __restrict__ Wq,
                                                 const float* __restrict__ bias,
                                                 float* __restrict__ out,
                                                 int Md, int Nd, int Kd) {
    __shared__ __align__(16) uint16_t As[2][BM * BK];   // 2 x 16 KB
    __shared__ __align__(16) uint16_t Bs[2][BN * BK];   // 2 x  8 KB

    const int tid  = threadIdx.x;
    const int wave = tid >> 5;
    const int lane = tid & 31;
    const int lh   = lane & 15;   // lane within half
    const int hi   = lane >> 4;   // which 16-lane half
    const int wm   = wave & 3;    // wave M coord (0..3) -> 64 rows each
    const int wn   = wave >> 2;   // wave N coord (0..1) -> 64 cols each
    const int m0   = blockIdx.y * BM;
    const int n0   = blockIdx.x * BN;

    // Accumulators: C element at (VGPR v, lane) is row v + 8*hi, col lh.
    // bias is per-column -> broadcast bias[col] into all 8 slots.
    v8f acc[4][4];
#pragma unroll
    for (int nt = 0; nt < 4; ++nt) {
        const float bv = bias[n0 + wn * 64 + nt * 16 + lh];
#pragma unroll
        for (int mt = 0; mt < 4; ++mt)
#pragma unroll
            for (int v = 0; v < 8; ++v) acc[mt][nt][v] = bv;
    }

    const int NK = Kd / BK;

    // Prologue: stage K-slice 0 into buffer 0.
    stage_tiles(Xh, Wq, As[0], Bs[0], m0, n0, 0, Kd, tid);
    wait_async_copies();
    __syncthreads();

    for (int kt = 0; kt < NK; ++kt) {
        const int cur = kt & 1;
        // Prefetch next K-slice into the other buffer (read-safe: all waves
        // finished reading it before the barrier that ended iteration kt-1).
        if (kt + 1 < NK)
            stage_tiles(Xh, Wq, As[cur ^ 1], Bs[cur ^ 1],
                        m0, n0, (kt + 1) * BK, Kd, tid);

        // B fragments (32x16 per tile): lane half 0 holds K=0..15 of col lh,
        // half 1 holds K=16..31 (contiguous k order, same as A).
        Frag bfrag[4];
#pragma unroll
        for (int nt = 0; nt < 4; ++nt) {
            const uint16_t* bp = Bs[cur] + (wn * 64 + nt * 16 + lh) * BK + hi * 16;
            bfrag[nt].u[0] = *(const v4u*)(bp);
            bfrag[nt].u[1] = *(const v4u*)(bp + 8);
        }

#pragma unroll
        for (int mt = 0; mt < 4; ++mt) {
            // A fragment (16x32): row M=lh; half 0 holds K=0..15,
            // half 1 holds K=16..31 (same k order as B).
            Frag af;
            const uint16_t* ap = As[cur] + (wm * 64 + mt * 16 + lh) * BK + hi * 16;
            af.u[0] = *(const v4u*)(ap);
            af.u[1] = *(const v4u*)(ap + 8);
#pragma unroll
            for (int nt = 0; nt < 4; ++nt) {
                acc[mt][nt] = __builtin_amdgcn_wmma_f32_16x16x32_bf16(
                    false, af.v, false, bfrag[nt].v,
                    (short)0, acc[mt][nt], false, false);
            }
        }

        wait_async_copies();   // async writes to buf[next] complete
        __syncthreads();       // everyone done reading buf[cur] / writing next
    }

    // Store: row = base + v + 8*hi, col = base + lh (coalesced across lanes)
#pragma unroll
    for (int mt = 0; mt < 4; ++mt)
#pragma unroll
        for (int nt = 0; nt < 4; ++nt) {
            const int mrow = m0 + wm * 64 + mt * 16 + hi * 8;
            const int ncol = n0 + wn * 64 + nt * 16 + lh;
            float* op = out + (size_t)mrow * Nd + ncol;
#pragma unroll
            for (int v = 0; v < 8; ++v)
                op[(size_t)v * Nd] = acc[mt][nt][v];
        }
}

// ---------------------------------------------------------------------------
// Launch
// ---------------------------------------------------------------------------
extern "C" void kernel_launch(void* const* d_in, const int* in_sizes, int n_in,
                              void* d_out, int out_size, void* d_ws, size_t ws_size,
                              hipStream_t stream) {
    (void)in_sizes; (void)n_in; (void)out_size; (void)ws_size;

    const int M = 8192;   // B*S = 4*2048
    const int K = 2048;
    const int N = 2048;

    const float* x    = (const float*)d_in[0];
    const float* w    = (const float*)d_in[1];
    const float* bias = (const float*)d_in[2];
    float* out        = (float*)d_out;

    uint16_t* Xh = (uint16_t*)d_ws;                                   // 32 MiB
    uint16_t* Wq = (uint16_t*)((char*)d_ws + (size_t)M * K * 2);      //  8 MiB

    const long long n4 = (long long)M * K / 4;
    convert_x<<<(int)((n4 + 255) / 256), 256, 0, stream>>>(x, Xh, n4);
    quantize_w<<<N, 256, 0, stream>>>(w, Wq, K);

    dim3 grid(N / BN, M / BM);   // 16 x 32
    gemm_wmma<<<grid, 256, 0, stream>>>(Xh, Wq, bias, out, M, N, K);
}
